// ContentBasedAttention_52725018525835
// MI455X (gfx1250) — compile-verified, matrix operand order checked by reading, not run on hardware
//
#include <hip/hip_runtime.h>
#include <hip/hip_bf16.h>

// Problem constants (from reference): B=32, S=4096, H=512
#define BATCH   32
#define SEQ     4096
#define HID     512
#define M_TILE  64                  // S-rows per workgroup in the energy kernel
#define KPAD    8                   // f16 elements of row padding in LDS (16B -> bank-conflict-free)
#define LDSTR   (HID + KPAD)        // 520 f16 elements per LDS row

typedef _Float16 f16x16 __attribute__((ext_vector_type(16)));
typedef float    f32x8  __attribute__((ext_vector_type(8)));

struct alignas(16) U4 { unsigned int a, b, c, d; };
union  Frag { f16x16 v; U4 u[2]; };
union  Acc  { f32x8  v; float f[8]; };
struct alignas(8) H4 { _Float16 x, y, z, w; };

// gfx1250 has a hardware V_TANH_F32 transcendental; use it when the builtin exists.
static __device__ inline float fast_tanh(float x) {
#if __has_builtin(__builtin_amdgcn_tanhf)
    return __builtin_amdgcn_tanhf(x);
#elif __has_builtin(__builtin_amdgcn_tanh_f32)
    return __builtin_amdgcn_tanh_f32(x);
#else
    return tanhf(x);
#endif
}

// ---------------------------------------------------------------------------
// Prep: W_s (fp32 [H,H], row = output col n, contiguous in k) -> f16
// ---------------------------------------------------------------------------
__global__ __launch_bounds__(256) void convert_ws_kernel(
    const float* __restrict__ Ws, _Float16* __restrict__ out)
{
    int i = (blockIdx.x * 256 + threadIdx.x) * 4;   // 4 elements per thread
    float4 f = *reinterpret_cast<const float4*>(Ws + i);
    H4 h;
    h.x = (_Float16)f.x; h.y = (_Float16)f.y;
    h.z = (_Float16)f.z; h.w = (_Float16)f.w;
    *reinterpret_cast<H4*>(out + i) = h;
}

// ---------------------------------------------------------------------------
// Prep: qb[b,n] = sum_k decoder_h[b,k] * W_h[n,k] + bias[n]   (tiny GEMV)
// ---------------------------------------------------------------------------
__global__ __launch_bounds__(512) void qb_kernel(
    const float* __restrict__ dh, const float* __restrict__ Wh,
    const float* __restrict__ bias, float* __restrict__ qb)
{
    int b = blockIdx.x;
    int n = threadIdx.x;                     // 512 threads, one output each
    const float* __restrict__ w = Wh + (size_t)n * HID;
    const float* __restrict__ q = dh + (size_t)b * HID;
    float acc = bias[n];
    #pragma unroll 8
    for (int k = 0; k < HID; ++k) acc = fmaf(q[k], w[k], acc);
    qb[b * HID + n] = acc;
}

// ---------------------------------------------------------------------------
// Main WMMA kernel:
//   energies[b,s] = sum_n  tanh( (keys @ Ws^T)[s,n] + qb[b,n] ) * v[n]
// Workgroup = 8 waves, tile = 64 S-rows x 512 cols, K = 512 (f16 WMMA, f32 acc).
// K is the outer loop: each k-step loads 4 A fragments (LDS) and 4 B fragments
// (global, L2-resident W_s) up front in one batch, then issues 16 WMMAs
// (4 col-strips x 4 row-blocks) with all 16 accumulators (128 VGPRs) live.
// ---------------------------------------------------------------------------
__global__ __launch_bounds__(256) void energy_kernel(
    const float*    __restrict__ keys,     // [B,S,H] fp32
    const _Float16* __restrict__ ws16,     // [H,H]  f16 (row n, contiguous k)
    const float*    __restrict__ qb,       // [B,H]
    const float*    __restrict__ vvec,     // [H]
    float*          __restrict__ energies) // [B,S]
{
    __shared__ _Float16 lds_keys[M_TILE * LDSTR];   // 66,560 B
    __shared__ float    lds_energy[M_TILE];

    const int b      = blockIdx.y;
    const int s_base = blockIdx.x * M_TILE;
    const int tid    = threadIdx.x;

    if (tid < M_TILE) lds_energy[tid] = 0.0f;

    // --- stage keys tile fp32 -> f16 in LDS (64 x 512, padded rows) ---
    {
        const size_t gbase = ((size_t)b * SEQ + s_base) * HID;
        #pragma unroll
        for (int it = 0; it < (M_TILE * HID / 4) / 256; ++it) {
            int id  = it * 256 + tid;           // chunk of 4 floats
            int row = id >> 7;                  // / (HID/4)
            int c4  = (id & 127) << 2;
            float4 f = *reinterpret_cast<const float4*>(keys + gbase + (size_t)row * HID + c4);
            H4 h;
            h.x = (_Float16)f.x; h.y = (_Float16)f.y;
            h.z = (_Float16)f.z; h.w = (_Float16)f.w;
            *reinterpret_cast<H4*>(&lds_keys[row * LDSTR + c4]) = h;
        }
    }
    __syncthreads();

    const int wave  = tid >> 5;
    const int lane  = tid & 31;
    const int lhalf = lane >> 4;         // 0: lanes 0-15, 1: lanes 16-31
    const int l16   = lane & 15;
    const int koff0 = lhalf ? 8  : 0;    // per-lane K chunk offsets (16B each)
    const int koff1 = lhalf ? 24 : 16;

    // Per-strip W_s row base for this lane (column n = no*128 + wave*16 + l16).
    // Strips are 128 rows apart -> compiler folds the distance into the
    // global-load immediate offset, so one base address serves all strips.
    const _Float16* __restrict__ wsbase =
        ws16 + (size_t)(wave * 16 + l16) * HID;

    Acc acc[4][4];                       // [col-strip][row-block]
    #pragma unroll
    for (int no = 0; no < 4; ++no)
        #pragma unroll
        for (int mb = 0; mb < 4; ++mb) acc[no][mb].v = (f32x8)0.0f;

    #pragma unroll 2
    for (int kk = 0; kk < HID / 32; ++kk) {
        const int kb = kk * 32;

        // Batch-load: 4 A fragments (LDS) + 4 B fragments (global) up front,
        // so the memory system can service all 16 requests while WMMAs drain
        // them with staggered waits.
        Frag af[4];
        #pragma unroll
        for (int mb = 0; mb < 4; ++mb) {
            const _Float16* arow = &lds_keys[(mb * 16 + l16) * LDSTR + kb];
            af[mb].u[0] = *reinterpret_cast<const U4*>(arow + koff0);
            af[mb].u[1] = *reinterpret_cast<const U4*>(arow + koff1);
        }
        Frag bf[4];
        #pragma unroll
        for (int no = 0; no < 4; ++no) {
            const _Float16* brow = wsbase + (size_t)(no * 128) * HID + kb;
            bf[no].u[0] = *reinterpret_cast<const U4*>(brow + koff0);
            bf[no].u[1] = *reinterpret_cast<const U4*>(brow + koff1);
        }

        #pragma unroll
        for (int no = 0; no < 4; ++no)
            #pragma unroll
            for (int mb = 0; mb < 4; ++mb)
                acc[no][mb].v = __builtin_amdgcn_wmma_f32_16x16x32_f16(
                    false, af[mb].v, false, bf[no].v, (short)0,
                    acc[no][mb].v, false, false);
    }

    // Epilogue: tanh(acc + qb[n]) * v[n], summed over this lane's columns
    float epart[4][8];
    #pragma unroll
    for (int mb = 0; mb < 4; ++mb)
        #pragma unroll
        for (int r = 0; r < 8; ++r) epart[mb][r] = 0.0f;

    #pragma unroll
    for (int no = 0; no < 4; ++no) {
        const int   ncol = no * 128 + wave * 16 + l16;
        const float qv   = qb[b * HID + ncol];
        const float vv   = vvec[ncol];
        #pragma unroll
        for (int mb = 0; mb < 4; ++mb)
            #pragma unroll
            for (int r = 0; r < 8; ++r)
                epart[mb][r] += fast_tanh(acc[no][mb].f[r] + qv) * vv;
    }

    // cross-lane / cross-wave reduction over the 512 cols via LDS atomics
    #pragma unroll
    for (int mb = 0; mb < 4; ++mb)
        #pragma unroll
        for (int r = 0; r < 8; ++r) {
            int m = mb * 16 + r + 8 * lhalf;    // C/D layout: VGPR r holds rows r / r+8
            atomicAdd(&lds_energy[m], epart[mb][r]);
        }
    __syncthreads();

    if (tid < M_TILE)
        energies[(size_t)b * SEQ + s_base + tid] = lds_energy[tid];
}

// ---------------------------------------------------------------------------
// Softmax over S per batch (in place: energies -> weights)
// ---------------------------------------------------------------------------
__global__ __launch_bounds__(256) void softmax_kernel(float* __restrict__ e)
{
    __shared__ float red[256];
    const int b = blockIdx.x, tid = threadIdx.x;
    float vals[SEQ / 256];
    float m = -__builtin_inff();
    #pragma unroll
    for (int i = 0; i < SEQ / 256; ++i) {
        vals[i] = e[(size_t)b * SEQ + i * 256 + tid];
        m = fmaxf(m, vals[i]);
    }
    red[tid] = m; __syncthreads();
    for (int s = 128; s > 0; s >>= 1) {
        if (tid < s) red[tid] = fmaxf(red[tid], red[tid + s]);
        __syncthreads();
    }
    m = red[0]; __syncthreads();

    float sum = 0.0f;
    #pragma unroll
    for (int i = 0; i < SEQ / 256; ++i) {
        vals[i] = __expf(vals[i] - m);
        sum += vals[i];
    }
    red[tid] = sum; __syncthreads();
    for (int s = 128; s > 0; s >>= 1) {
        if (tid < s) red[tid] += red[tid + s];
        __syncthreads();
    }
    const float inv = 1.0f / red[0];
    #pragma unroll
    for (int i = 0; i < SEQ / 256; ++i)
        e[(size_t)b * SEQ + i * 256 + tid] = vals[i] * inv;
}

// ---------------------------------------------------------------------------
// context[b,n] = sum_s weights[b,s] * keys[b,s,n]   (second keys pass)
// ---------------------------------------------------------------------------
__global__ __launch_bounds__(256) void zero_out_kernel(float* __restrict__ out)
{
    out[blockIdx.x * 256 + threadIdx.x] = 0.0f;
}

__global__ __launch_bounds__(256) void context_kernel(
    const float* __restrict__ keys, const float* __restrict__ weights,
    float* __restrict__ out)
{
    __shared__ float wsm[512];
    const int b = blockIdx.y, sc = blockIdx.x * 512, tid = threadIdx.x;
    wsm[tid]       = weights[(size_t)b * SEQ + sc + tid];
    wsm[tid + 256] = weights[(size_t)b * SEQ + sc + tid + 256];
    __syncthreads();

    const int c0 = tid * 2;
    float ax = 0.0f, ay = 0.0f;
    const float* __restrict__ kb = keys + ((size_t)b * SEQ + sc) * HID;
    #pragma unroll 4
    for (int s = 0; s < 512; ++s) {
        float  w  = wsm[s];
        float2 k2 = *reinterpret_cast<const float2*>(kb + (size_t)s * HID + c0);
        ax = fmaf(w, k2.x, ax);
        ay = fmaf(w, k2.y, ay);
    }
    atomicAdd(&out[b * HID + c0],     ax);
    atomicAdd(&out[b * HID + c0 + 1], ay);
}

// ---------------------------------------------------------------------------
extern "C" void kernel_launch(void* const* d_in, const int* in_sizes, int n_in,
                              void* d_out, int out_size, void* d_ws, size_t ws_size,
                              hipStream_t stream)
{
    (void)in_sizes; (void)n_in; (void)out_size; (void)ws_size;

    const float* keys = (const float*)d_in[0];   // [B,S,H]
    const float* dh   = (const float*)d_in[1];   // [1,B,H]
    const float* Wh   = (const float*)d_in[2];   // [H,H]
    const float* Ws   = (const float*)d_in[3];   // [H,H]
    const float* v    = (const float*)d_in[4];   // [H]
    const float* bias = (const float*)d_in[5];   // [H]
    float* out        = (float*)d_out;           // [B,1,H]

    char* w = (char*)d_ws;
    _Float16* ws16     = (_Float16*)w;                               // 512 KB
    float*    qb       = (float*)(w + (size_t)HID * HID * 2);        // 64 KB
    float*    energies = (float*)(w + (size_t)HID * HID * 2
                                    + (size_t)BATCH * HID * 4);      // 512 KB

    convert_ws_kernel<<<dim3(HID * HID / 1024), dim3(256), 0, stream>>>(Ws, ws16);
    qb_kernel        <<<dim3(BATCH),            dim3(512), 0, stream>>>(dh, Wh, bias, qb);
    energy_kernel    <<<dim3(SEQ / M_TILE, BATCH), dim3(256), 0, stream>>>(
                        keys, ws16, qb, v, energies);
    softmax_kernel   <<<dim3(BATCH),            dim3(256), 0, stream>>>(energies);
    zero_out_kernel  <<<dim3(BATCH * HID / 256), dim3(256), 0, stream>>>(out);
    context_kernel   <<<dim3(SEQ / 512, BATCH), dim3(256), 0, stream>>>(keys, energies, out);
}
